// AgentAttention_27023934226836
// MI455X (gfx1250) — matrix operand order repaired and checked by reading, not compiled, not run on hardware
//
#include <hip/hip_runtime.h>
#include <hip/hip_bf16.h>
#include <cstdint>
#include <cstddef>

// ---------------- problem constants (from reference setup) ----------------
#define BATCH   32
#define NTOK    1176
#define CDIM    768
#define NHEAD   12
#define HD      64
#define NMT     392      // 2*14*14 merged tokens
#define NS      784      // 28*28 spatial tokens
#define AG      49       // agents (7x7)
#define NPAD    1184     // token dim padded to multiple of 32
#define BHTOT   (BATCH*NHEAD)
#define MROWS   (BATCH*NTOK) // 37632, divisible by 128

// ---------------- WMMA types / helpers ----------------
typedef __attribute__((ext_vector_type(16))) __bf16 v16bf;
typedef __attribute__((ext_vector_type(8)))  float  v8f;

union FragBF { v16bf v; unsigned int u[8]; };

__device__ inline unsigned short f2bf(float f) {
    unsigned int x = __float_as_uint(f);
    return (unsigned short)((x + 0x7FFFu + ((x >> 16) & 1u)) >> 16); // RNE
}
__device__ inline float bf2f(unsigned short h) {
    return __uint_as_float(((unsigned int)h) << 16);
}
__device__ inline unsigned int pack2(float lo, float hi) {
    return (unsigned int)f2bf(lo) | ((unsigned int)f2bf(hi) << 16);
}
__device__ inline v8f wmma_bf16(const FragBF& a, const FragBF& b, v8f c) {
    return __builtin_amdgcn_wmma_f32_16x16x32_bf16(false, a.v, false, b.v,
                                                   (short)0, c, false, false);
}
__device__ inline v8f vzero8() {
    v8f z;
#pragma unroll
    for (int r = 0; r < 8; ++r) z[r] = 0.0f;
    return z;
}

// ---- fragment builders matching CDNA5 ISA VGPR layouts (CDNA5 ISA §7.12.2) ----
// A (16x32 bf16): lane m (and m+16) holds row m; lanes<16: K 0..7,16..23; lanes>=16: K 8..15,24..31
__device__ inline void fragA_rows(FragBF& a, const unsigned short* base, int rbase,
                                  int ks, int lane) {
    const unsigned int* u = (const unsigned int*)base; // rows of 64 bf16 = 32 uints
    int ln = lane & 15, hi = lane >> 4;
    size_t row = (size_t)(rbase + ln);
#pragma unroll
    for (int v = 0; v < 8; ++v) {
        int p = ks * 16 + ((v >> 2) << 3) + (hi << 2) + (v & 3);
        a.u[v] = u[row * 32 + p];
    }
}
// B (32x16 bf16) where B[k=d][n=tok] and source is tok-major rows [tok][64]
__device__ inline void fragB_rowsT(FragBF& b, const unsigned short* base, int nbase,
                                   int ks, int lane) {
    const unsigned int* u = (const unsigned int*)base;
    int ln = lane & 15, hi = lane >> 4;
    size_t col = (size_t)(nbase + ln);
#pragma unroll
    for (int v = 0; v < 8; ++v)
        b.u[v] = u[col * 32 + ks * 16 + ((v >> 2) << 3) + (hi << 2) + (v & 3)];
}
// B (32x16 bf16) where B[k=t][n=d], source row-major [t][64] (gather column d)
__device__ inline void fragB_cols(FragBF& b, const unsigned short* base, int tbase,
                                  int dbase, int lane) {
    int ln = lane & 15, hi = lane >> 4;
    int d = dbase + ln;
    int t0 = tbase + hi * 8;
#pragma unroll
    for (int v = 0; v < 8; ++v) {
        unsigned int lo = base[(size_t)(t0 + ((v >> 2) << 4) + 2 * (v & 3)) * 64 + d];
        unsigned int hi16 = base[(size_t)(t0 + ((v >> 2) << 4) + 2 * (v & 3) + 1) * 64 + d];
        b.u[v] = lo | (hi16 << 16);
    }
}
// A (16x32 bf16) from LDS P-matrix [16][rowStrideU uints]
__device__ inline void fragA_lds(FragBF& a, const unsigned int* S, int rowStrideU,
                                 int tb, int lane) {
    int ln = lane & 15, hi = lane >> 4;
#pragma unroll
    for (int v = 0; v < 8; ++v) {
        int p = (tb >> 1) + ((v >> 2) << 3) + (hi << 2) + (v & 3);
        a.u[v] = S[ln * rowStrideU + p];
    }
}

// ---------------- f32 -> bf16 bulk conversion (one pass; keeps GEMM loop pure) ----
__global__ __launch_bounds__(256) void cvt_bf16_kernel(const float* __restrict__ src,
                                                       unsigned short* __restrict__ dst,
                                                       int n4) {
    int i = blockIdx.x * 256 + threadIdx.x;
    if (i < n4) {
        float4 d = ((const float4*)src)[i];
        uint2 o;
        o.x = pack2(d.x, d.y);
        o.y = pack2(d.z, d.w);
        ((uint2*)dst)[i] = o;
    }
}

// ---------------- GEMM: Y = A @ W^T + bias (all-bf16 inputs, f32 accum) ----------
// A: MROWS x K bf16 row-major, W: Ncols x K bf16 row-major.
// 128x128 block, 4 waves, each wave owns a 64x64 quadrant (4x4 WMMA frags).
template <int EPI> // EPI 0: scatter to Q/K/V, 1: f32 out
__global__ __launch_bounds__(128) void gemm_wmma_kernel(
    const unsigned short* __restrict__ Ab, const unsigned short* __restrict__ Wb,
    const float* __restrict__ bias, int K, int Ncols,
    unsigned short* __restrict__ Qb, unsigned short* __restrict__ Kb,
    unsigned short* __restrict__ Vb, float* __restrict__ Out) {
    const int tid = threadIdx.x;
    const int lane = tid & 31, wv = tid >> 5;
    const int wr = wv >> 1, wc = wv & 1;
    const int n0 = blockIdx.x * 128, m0 = blockIdx.y * 128;
    __shared__ unsigned int As[128 * 17]; // 16 bf16-pairs per row + 1 pad
    __shared__ unsigned int Bs[128 * 17];

    v8f acc[4][4];
#pragma unroll
    for (int i = 0; i < 4; ++i)
#pragma unroll
        for (int j = 0; j < 4; ++j) acc[i][j] = vzero8();

    const int ln = lane & 15, hi = lane >> 4;
    const int lrow = tid >> 2, lq = tid & 3; // tile loader coords (512 uint4 / 128 thr)

    for (int k0 = 0; k0 < K; k0 += 32) {
        // prefetch next k-tile rows handled by this thread (global_prefetch_b8 path)
        if (k0 + 32 < K) {
            __builtin_prefetch(Ab + (size_t)(m0 + lrow) * K + (k0 + 32) + lq * 8, 0, 1);
            __builtin_prefetch(Wb + (size_t)(n0 + lrow) * K + (k0 + 32) + lq * 8, 0, 1);
        }
        __syncthreads();
#pragma unroll
        for (int i = 0; i < 4; ++i) {
            int row = lrow + i * 32;
            uint4 da = *((const uint4*)(Ab + (size_t)(m0 + row) * K + k0) + lq);
            As[row * 17 + lq * 4 + 0] = da.x;
            As[row * 17 + lq * 4 + 1] = da.y;
            As[row * 17 + lq * 4 + 2] = da.z;
            As[row * 17 + lq * 4 + 3] = da.w;
            uint4 db = *((const uint4*)(Wb + (size_t)(n0 + row) * K + k0) + lq);
            Bs[row * 17 + lq * 4 + 0] = db.x;
            Bs[row * 17 + lq * 4 + 1] = db.y;
            Bs[row * 17 + lq * 4 + 2] = db.z;
            Bs[row * 17 + lq * 4 + 3] = db.w;
        }
        __syncthreads();

        FragBF a[4], b[4];
#pragma unroll
        for (int i = 0; i < 4; ++i) {
            int row = wr * 64 + i * 16 + ln;
#pragma unroll
            for (int v = 0; v < 8; ++v) {
                int p = ((v >> 2) << 3) + (hi << 2) + (v & 3);
                a[i].u[v] = As[row * 17 + p];
            }
        }
#pragma unroll
        for (int j = 0; j < 4; ++j) {
            int col = wc * 64 + j * 16 + ln;
#pragma unroll
            for (int v = 0; v < 8; ++v) b[j].u[v] = Bs[col * 17 + ((v >> 2) << 3) + (hi << 2) + (v & 3)];
        }
#pragma unroll
        for (int i = 0; i < 4; ++i)
#pragma unroll
            for (int j = 0; j < 4; ++j) acc[i][j] = wmma_bf16(a[i], b[j], acc[i][j]);
    }

    // epilogue: hoist per-column (j) and per-row (i,r) index math
    int nj[4], whichj[4], hj[4], dj[4];
    float biasj[4];
#pragma unroll
    for (int j = 0; j < 4; ++j) {
        int n = n0 + wc * 64 + j * 16 + ln;
        nj[j] = n;
        biasj[j] = bias[n];
        if (EPI == 0) {
            int which = n / CDIM;
            int c = n - which * CDIM;
            whichj[j] = which;
            hj[j] = c >> 6;
            dj[j] = c & 63;
        }
    }
#pragma unroll
    for (int i = 0; i < 4; ++i)
#pragma unroll
        for (int r = 0; r < 8; ++r) {
            int m = m0 + wr * 64 + i * 16 + hi * 8 + r;
            int bidx = m / NTOK, tok = m - bidx * NTOK; // only used for EPI==0
#pragma unroll
            for (int j = 0; j < 4; ++j) {
                float val = acc[i][j][r] + biasj[j];
                if (EPI == 0) {
                    unsigned short* dst =
                        (whichj[j] == 0) ? Qb : (whichj[j] == 1) ? Kb : Vb;
                    dst[(((size_t)bidx * NHEAD + hj[j]) * NPAD + tok) * 64 + dj[j]] =
                        f2bf(val);
                } else {
                    Out[(size_t)m * Ncols + nj[j]] = val;
                }
            }
        }
}

// ---------------- zero padded token tails of Q/K/V ----------------
__global__ void init_tails_kernel(unsigned short* Qb, unsigned short* Kb,
                                  unsigned short* Vb) {
    int i = blockIdx.x * 256 + threadIdx.x; // BHTOT * 8 rows * 64
    if (i < BHTOT * 8 * 64) {
        int bh = i / (8 * 64);
        int r = i - bh * 512;
        size_t off = ((size_t)bh * NPAD + NTOK) * 64 + r;
        Qb[off] = 0; Kb[off] = 0; Vb[off] = 0;
    }
}

// ---------------- agent pooling (reproduces reference reshape quirk) ----------------
// q_s flat (h,n,d) reinterpreted as (y,x,c): f = y*21504 + x*768 + c
__global__ void pool_kernel(const unsigned short* __restrict__ Qb,
                            unsigned short* __restrict__ Ag) {
    int bh = blockIdx.x;
    int b = bh / NHEAD, h = bh - b * NHEAD;
    for (int i = threadIdx.x; i < 64 * 64; i += blockDim.x) {
        int a = i >> 6, d = i & 63;
        unsigned short out = 0;
        if (a < AG) {
            int ay = a / 7, ax = a - ay * 7;
            int c = h * 64 + d;
            float s = 0.f;
#pragma unroll
            for (int yy = 0; yy < 4; ++yy)
#pragma unroll
                for (int xx = 0; xx < 4; ++xx) {
                    int y = ay * 4 + yy, x = ax * 4 + xx;
                    int f = y * 21504 + x * 768 + c;
                    int hp = f / 50176;
                    int rem = f - hp * 50176;
                    int n = rem >> 6, dd = rem & 63;
                    s += bf2f(Qb[(((size_t)b * NHEAD + hp) * NPAD + NMT + n) * 64 + dd]);
                }
            out = f2bf(s * (1.f / 16.f));
        }
        Ag[((size_t)bh * 64 + a) * 64 + d] = out;
    }
}

// ---------------- fused attention tile: S = scale*Q.K^T, softmax, O = P.V ----------------
// MODE 0: mt self-attn (T=392) -> xcat rows [0,392)
// MODE 1: agent attn   (T=1176) -> agent_v (64 rows, zero a>=49)
// MODE 2: q_s over agents (T=49) -> xcat rows [392,1176)
template <int MODE>
__global__ __launch_bounds__(128) void attn_kernel(
    const unsigned short* __restrict__ Qb, const unsigned short* __restrict__ Kb,
    const unsigned short* __restrict__ Vb, const unsigned short* __restrict__ Ag,
    const unsigned short* __restrict__ Av, unsigned short* __restrict__ AvOut,
    unsigned short* __restrict__ xcat) {
    constexpr int TV = (MODE == 0) ? NMT : (MODE == 1) ? NTOK : AG;
    constexpr int TP = (MODE == 0) ? 416 : (MODE == 1) ? NPAD : 64;
    constexpr int ROWU = (MODE == 0) ? 209 : (MODE == 1) ? 593 : 33; // odd uint stride
    constexpr int CH = (TV + 7) / 8;

    const int bh = blockIdx.x;
    const int qt = blockIdx.y;
    const int tid = threadIdx.x, lane = tid & 31, wv = tid >> 5;
    const int ln = lane & 15, hi = lane >> 4;

    __shared__ unsigned int Sl[16 * ROWU];
    __shared__ float red[16 * 8];
    __shared__ float rinv[16];
    unsigned short* S16 = (unsigned short*)Sl;

    const unsigned short* Arow;
    const unsigned short* Ksrc;
    const unsigned short* Vsrc;
    int rbase;
    if (MODE == 0) {
        Arow = Qb + (size_t)bh * NPAD * 64; rbase = qt * 16;
        Ksrc = Kb + (size_t)bh * NPAD * 64; Vsrc = Vb + (size_t)bh * NPAD * 64;
    } else if (MODE == 1) {
        Arow = Ag + (size_t)bh * 64 * 64; rbase = qt * 16;
        Ksrc = Kb + (size_t)bh * NPAD * 64; Vsrc = Vb + (size_t)bh * NPAD * 64;
    } else {
        Arow = Qb + (size_t)bh * NPAD * 64; rbase = NMT + qt * 16;
        Ksrc = Ag + (size_t)bh * 64 * 64; Vsrc = Av + (size_t)bh * 64 * 64;
    }

    // -------- phase 1: S tiles (scale = hd^-0.5 = 0.125) --------
    FragBF aq0, aq1;
    fragA_rows(aq0, Arow, rbase, 0, lane);
    fragA_rows(aq1, Arow, rbase, 1, lane);
    for (int t = wv * 16; t < TP; t += 64) {
        FragBF b0, b1;
        fragB_rowsT(b0, Ksrc, t, 0, lane);
        fragB_rowsT(b1, Ksrc, t, 1, lane);
        v8f c = vzero8();
        c = wmma_bf16(aq0, b0, c);
        c = wmma_bf16(aq1, b1, c);
#pragma unroll
        for (int r = 0; r < 8; ++r) {
            int m = hi * 8 + r;
            S16[m * (ROWU * 2) + t + ln] = f2bf(c[r] * 0.125f);
        }
    }
    __syncthreads();

    // zero pad columns [TV,TP) so padded K/V never contributes
    for (int i = tid; i < 16 * (TP - TV); i += 128) {
        int r = i / (TP - TV), cc = TV + i % (TP - TV);
        S16[r * (ROWU * 2) + cc] = 0;
    }

    // -------- phase 2: softmax (unnormalized exp; 1/sum folded into epilogue) --------
    {
        int r = tid & 15, ck = tid >> 4;
        int cbeg = ck * CH;
        int cend = (cbeg + CH < TV) ? (cbeg + CH) : TV;
        float mx = -3.0e38f;
        for (int cc = cbeg; cc < cend; ++cc)
            mx = fmaxf(mx, bf2f(S16[r * (ROWU * 2) + cc]));
        red[r * 8 + ck] = mx;
        __syncthreads();
        float rmx = red[r * 8 + 0];
#pragma unroll
        for (int k = 1; k < 8; ++k) rmx = fmaxf(rmx, red[r * 8 + k]);
        __syncthreads();
        float s = 0.f;
        for (int cc = cbeg; cc < cend; ++cc) {
            float e = __expf(bf2f(S16[r * (ROWU * 2) + cc]) - rmx);
            S16[r * (ROWU * 2) + cc] = f2bf(e);
            s += e;
        }
        red[r * 8 + ck] = s;
        __syncthreads();
        if (ck == 0) {
            float ts = 0.f;
#pragma unroll
            for (int k = 0; k < 8; ++k) ts += red[r * 8 + k];
            rinv[r] = 1.0f / ts;
        }
    }
    __syncthreads();

    // -------- phase 3: O = P @ V ; wave wv owns d-columns [wv*16, wv*16+16) --------
    v8f o = vzero8();
    const int dg = wv * 16;
    for (int tb = 0; tb < TP; tb += 32) {
        FragBF ap, bv;
        fragA_lds(ap, Sl, ROWU, tb, lane);
        fragB_cols(bv, Vsrc, tb, dg, lane);
        o = wmma_bf16(ap, bv, o);
    }
#pragma unroll
    for (int r = 0; r < 8; ++r) {
        int m = hi * 8 + r;
        int d = dg + ln;
        float val = o[r] * rinv[m];
        if (MODE == 0) {
            int tok = qt * 16 + m;
            if (tok < NMT) {
                int b = bh / NHEAD, h = bh - (bh / NHEAD) * NHEAD;
                xcat[((size_t)b * NTOK + tok) * CDIM + h * 64 + d] = f2bf(val);
            }
        } else if (MODE == 1) {
            int a = qt * 16 + m;
            AvOut[((size_t)bh * 64 + a) * 64 + d] = (a < AG) ? f2bf(val) : (unsigned short)0;
        } else {
            int tok = NMT + qt * 16 + m;
            int b = bh / NHEAD, h = bh - (bh / NHEAD) * NHEAD;
            xcat[((size_t)b * NTOK + tok) * CDIM + h * 64 + d] = f2bf(val);
        }
    }
}

// ---------------- host launch ----------------
extern "C" void kernel_launch(void* const* d_in, const int* in_sizes, int n_in,
                              void* d_out, int out_size, void* d_ws, size_t ws_size,
                              hipStream_t stream) {
    const float* x      = (const float*)d_in[0];
    const float* qkv_w  = (const float*)d_in[1];
    const float* qkv_b  = (const float*)d_in[2];
    const float* proj_w = (const float*)d_in[3];
    const float* proj_b = (const float*)d_in[4];
    // t_h/t_w/s_h/s_w (d_in[5..8]) are fixed by setup_inputs; baked as constants.

    char* ws = (char*)d_ws;
    size_t off = 0;
    auto carve = [&](size_t bytes) -> void* {
        void* p = ws + off;
        off = (off + bytes + 255) & ~(size_t)255;
        return p;
    };
    const size_t qkvE = (size_t)BHTOT * NPAD * 64;
    unsigned short* Qb  = (unsigned short*)carve(qkvE * 2);
    unsigned short* Kb  = (unsigned short*)carve(qkvE * 2);
    unsigned short* Vb  = (unsigned short*)carve(qkvE * 2);
    unsigned short* Agp = (unsigned short*)carve((size_t)BHTOT * 64 * 64 * 2);
    unsigned short* Avp = (unsigned short*)carve((size_t)BHTOT * 64 * 64 * 2);
    unsigned short* Xc  = (unsigned short*)carve((size_t)BATCH * NTOK * CDIM * 2);
    unsigned short* Xb  = (unsigned short*)carve((size_t)MROWS * CDIM * 2);
    unsigned short* Wq  = (unsigned short*)carve((size_t)3 * CDIM * CDIM * 2);
    unsigned short* Wp  = (unsigned short*)carve((size_t)CDIM * CDIM * 2);

    // 0) one-time f32 -> bf16 conversions (GEMM inner loops become pure copy+WMMA)
    {
        int n4x = MROWS * CDIM / 4;           // 7,225,344
        int n4q = 3 * CDIM * CDIM / 4;        // 442,368
        int n4p = CDIM * CDIM / 4;            // 147,456
        cvt_bf16_kernel<<<(n4x + 255) / 256, 256, 0, stream>>>(x, Xb, n4x);
        cvt_bf16_kernel<<<(n4q + 255) / 256, 256, 0, stream>>>(qkv_w, Wq, n4q);
        cvt_bf16_kernel<<<(n4p + 255) / 256, 256, 0, stream>>>(proj_w, Wp, n4p);
    }

    // 0b) zero padded token tails (avoid NaN poisoning through 0*garbage)
    init_tails_kernel<<<(BHTOT * 8 * 64 + 255) / 256, 256, 0, stream>>>(Qb, Kb, Vb);

    // 1) QKV GEMM: (37632 x 2304 x 768), scatter to Q/K/V bf16 [B][H][tok][64]
    gemm_wmma_kernel<0><<<dim3(2304 / 128, MROWS / 128), 128, 0, stream>>>(
        Xb, Wq, qkv_b, CDIM, 3 * CDIM, Qb, Kb, Vb, nullptr);

    // 2) agent pooling (head-mixing reshape semantics of the reference)
    pool_kernel<<<BHTOT, 256, 0, stream>>>(Qb, Agp);

    // 3) mt self-attention -> xcat[:,0:392)
    attn_kernel<0><<<dim3(BHTOT, 25), 128, 0, stream>>>(Qb, Kb, Vb, Agp, Avp, Avp, Xc);

    // 4) agent attention over all 1176 keys -> agent_v
    attn_kernel<1><<<dim3(BHTOT, 4), 128, 0, stream>>>(Qb, Kb, Vb, Agp, Avp, Avp, Xc);

    // 5) q_s attention over 49 agents -> xcat[:,392:1176)
    attn_kernel<2><<<dim3(BHTOT, 49), 128, 0, stream>>>(Qb, Kb, Vb, Agp, Avp, Avp, Xc);

    // 6) projection GEMM: (37632 x 768 x 768) -> f32 d_out
    gemm_wmma_kernel<1><<<dim3(CDIM / 128, MROWS / 128), 128, 0, stream>>>(
        Xc, Wp, proj_b, CDIM, CDIM, nullptr, nullptr, nullptr, (float*)d_out);
}
